// SelfAttention_45148696216798
// MI455X (gfx1250) — matrix operand order repaired and checked by reading, not compiled, not run on hardware
//
#include <hip/hip_runtime.h>
#include <hip/hip_bf16.h>

typedef __bf16 bf16;
typedef __attribute__((ext_vector_type(16))) __bf16 v16bf;
typedef __attribute__((ext_vector_type(8)))  __bf16 v8bf;
typedef __attribute__((ext_vector_type(4)))  __bf16 v4bf;
typedef __attribute__((ext_vector_type(8)))  float  v8f;

#define BATCH 8
#define NSEQ  2048
#define DDIM  512
#define SROW  2064   // padded score-row stride (floats): 2064 % 64 = 16 -> no bank clash between half-waves
#define LDS_BYTES (16 * SROW * 4 + 16 * DDIM * 2)

// ---------------- Kernel 1: fp32 -> bf16 conversion of x into workspace ----------------
__global__ void cvt_f32_bf16(const float* __restrict__ x, bf16* __restrict__ w, int n4) {
  int stride = gridDim.x * blockDim.x;
  for (int i = blockIdx.x * blockDim.x + threadIdx.x; i < n4; i += stride) {
    float4 v = ((const float4*)x)[i];
    v4bf o = { (bf16)v.x, (bf16)v.y, (bf16)v.z, (bf16)v.w };
    ((v4bf*)w)[i] = o;
  }
}

// ---------------- Kernel 2: fused scores (x x^T via WMMA bf16) + row softmax ----------------
__global__ __launch_bounds__(256) void attn_scores_softmax(
    const bf16* __restrict__ xw, float* __restrict__ out) {
  extern __shared__ char smem[];
  float* S = (float*)smem;                       // 16 x SROW fp32 score staging
  bf16*  A = (bf16*)(smem + 16 * SROW * 4);      // 16 x 512 bf16 A tile

  const int tid  = threadIdx.x;
  const int lane = tid & 31;
  const int wave = tid >> 5;
  const int blk  = blockIdx.x;                   // b*128 + row-block
  const int b    = blk >> 7;
  const int i0   = (blk & 127) << 4;             // first of 16 rows

  const bf16* xb = xw + (size_t)b * NSEQ * DDIM;

  // Phase 1: cooperatively stage A tile (16 rows x 512) into LDS, 128-bit chunks
  {
    const v8bf* src = (const v8bf*)(xb + (size_t)i0 * DDIM);  // 16*512/8 = 1024 chunks
    v8bf* dst = (v8bf*)A;
    for (int i = tid; i < (16 * DDIM) / 8; i += 256) dst[i] = src[i];
  }
  __syncthreads();

  const int half  = lane >> 4;     // 0: lanes 0-15, 1: lanes 16-31
  const int l16   = lane & 15;
  const int koffA = half * 8;      // A frag: K {0..7,16..23} vs {8..15,24..31}
  const int koffB = half * 8;      // B frag: K {0..7,16..23} vs {8..15,24..31}

  // Preload all 16 A fragments (K chunks of 32) into registers; reused for all column tiles
  v16bf afrag[16];
  #pragma unroll
  for (int kk = 0; kk < 16; ++kk) {
    const bf16* ap = A + l16 * DDIM + kk * 32 + koffA;
    union { v16bf v; v8bf h[2]; } u;
    u.h[0] = *(const v8bf*)(ap);
    u.h[1] = *(const v8bf*)(ap + 16);
    afrag[kk] = u.v;
  }

  // Phase 2: each wave computes 16 of the 128 column tiles (interleaved for load spread)
  for (int t = 0; t < 16; ++t) {
    const int jt = t * 8 + wave;
    const int j0 = jt * 16;
    const bf16* brow = xb + (size_t)(j0 + l16) * DDIM + koffB;
    v8f acc = {0.f, 0.f, 0.f, 0.f, 0.f, 0.f, 0.f, 0.f};
    #pragma unroll
    for (int kk = 0; kk < 16; ++kk) {
      union { v16bf v; v8bf h[2]; } ub;
      ub.h[0] = *(const v8bf*)(brow + kk * 32);
      ub.h[1] = *(const v8bf*)(brow + kk * 32 + 16);
      acc = __builtin_amdgcn_wmma_f32_16x16x32_bf16(false, afrag[kk], false, ub.v,
                                                    (short)0, acc, false, false);
    }
    // C/D layout: VGPR r -> M = r (lanes 0-15) or r+8 (lanes 16-31), N = lane & 15
    const int mb = half * 8;
    #pragma unroll
    for (int r = 0; r < 8; ++r)
      S[(mb + r) * SROW + j0 + l16] = acc[r];
  }
  __syncthreads();

  // Phase 3: row softmax; 2 rows per wave (8 waves x 2 = 16 rows)
  for (int rr = 0; rr < 2; ++rr) {
    const int row = wave * 2 + rr;
    float* srow = S + row * SROW;

    float m = -3.402823466e38f;
    for (int c = lane; c < NSEQ; c += 32) m = fmaxf(m, srow[c]);
    #pragma unroll
    for (int off = 16; off >= 1; off >>= 1) m = fmaxf(m, __shfl_xor(m, off, 32));

    float sum = 0.f;
    for (int c = lane; c < NSEQ; c += 32) {
      float e = __expf(srow[c] - m);
      srow[c] = e;
      sum += e;
    }
    #pragma unroll
    for (int off = 16; off >= 1; off >>= 1) sum += __shfl_xor(sum, off, 32);
    const float inv = 1.0f / sum;

    float4* orow = (float4*)(out + (size_t)(b * NSEQ + i0 + row) * NSEQ);
    const float4* s4 = (const float4*)srow;
    for (int c = lane; c < NSEQ / 4; c += 32) {
      float4 v = s4[c];
      v.x *= inv; v.y *= inv; v.z *= inv; v.w *= inv;
      orow[c] = v;
    }
  }
}

extern "C" void kernel_launch(void* const* d_in, const int* in_sizes, int n_in,
                              void* d_out, int out_size, void* d_ws, size_t ws_size,
                              hipStream_t stream) {
  (void)in_sizes; (void)n_in; (void)out_size; (void)ws_size;
  const float* x = (const float*)d_in[0];
  float* out = (float*)d_out;
  bf16* xw = (bf16*)d_ws;   // needs 8*2048*512*2 = 16 MiB of scratch

  const int n4 = (BATCH * NSEQ * DDIM) / 4;
  cvt_f32_bf16<<<2048, 256, 0, stream>>>(x, xw, n4);

  attn_scores_softmax<<<BATCH * (NSEQ / 16), 256, LDS_BYTES, stream>>>(xw, out);
}
